// SKVMN_21320217657937
// MI455X (gfx1250) — compile-verified, matrix operand order checked by reading, not run on hardware
//
#include <hip/hip_runtime.h>
#include <hip/hip_bf16.h>
#include <stdint.h>

// ---------------------------------------------------------------------------
// SKVMN on MI455X (gfx1250): one persistent kernel, 16 WGs x 16 batch rows.
// All per-step GEMMs use v_wmma_f32_16x16x32_f16 (M=16 batch tile per wave).
// ---------------------------------------------------------------------------

typedef __attribute__((ext_vector_type(16))) _Float16 v16h;
typedef __attribute__((ext_vector_type(8)))  _Float16 v8h;
typedef __attribute__((ext_vector_type(8)))  float    v8f;

#define B_TOT 256
#define S_LEN 200
#define D_DIM 128
#define M_MEM 64
#define NQ_C  5000
#define BT    16          // batch rows per workgroup (== WMMA M)
#define NTH   256         // 8 waves (wave32)

// ---- workspace layout (bytes, each region 256-aligned) --------------------
#define WS_MEM_B   ((size_t)B_TOT * M_MEM * D_DIM * 4)          // 8 MB f32
#define WS_FT_B    ((size_t)S_LEN * B_TOT * D_DIM * 2)          // 13.1 MB f16
#define WS_SIG_B   ((size_t)B_TOT * S_LEN * 4 * 4)              // 0.8 MB u32
#define WS_H_B     ((size_t)S_LEN * B_TOT * D_DIM * 2)          // 13.1 MB f16
#define N_MK16     (M_MEM * D_DIM)
#define N_WF16     (D_DIM * 2 * D_DIM)
#define N_WA16     (D_DIM * 2 * D_DIM)
#define N_WE16     (D_DIM * D_DIM)
#define N_WADD16   (D_DIM * D_DIM)
#define N_WIH16    (4 * D_DIM * D_DIM)
#define N_WHH16    (4 * D_DIM * D_DIM)

static __device__ __forceinline__ float sigm(float x) {
  return 1.0f / (1.0f + __expf(-x));
}

// A-fragment loader (16x32 f16 tile, row-major source, stride ld halves).
// ISA layout: lane<16 -> row=lane,   K in {0..7, 16..23}
//             lane>=16 -> row=lane-16, K in {8..15, 24..31}
// Same loader produces the B (KxN) fragment from W stored [out][in] row-major.
static __device__ __forceinline__ v16h frag_ld(const _Float16* src, int ld) {
  const int lane = threadIdx.x & 31;
  const _Float16* r = src + (lane & 15) * ld + ((lane & 16) ? 8 : 0);
  v8h lo = *(const v8h*)(r);
  v8h hi = *(const v8h*)(r + 16);
  v16h a;
#pragma unroll
  for (int i = 0; i < 8; ++i) { a[i] = lo[i]; a[8 + i] = hi[i]; }
  return a;
}

static __device__ __forceinline__ v8f wmma32(v16h a, v16h b, v8f c) {
  return __builtin_amdgcn_wmma_f32_16x16x32_f16(
      /*neg_a=*/false, a, /*neg_b=*/false, b,
      /*c_mod=*/(short)0, c, /*reuse_a=*/false, /*reuse_b=*/false);
}

// C = A[16,K] x (Wtile[16 rows of K])^T -> one 16x16 f32 accumulator tile
static __device__ __forceinline__ v8f gemm_tile(const _Float16* A, int lda,
                                                const _Float16* Wt, int ldw,
                                                int K) {
  v8f acc = {0.f, 0.f, 0.f, 0.f, 0.f, 0.f, 0.f, 0.f};
  for (int k = 0; k < K; k += 32)
    acc = wmma32(frag_ld(A + k, lda), frag_ld(Wt + k, ldw), acc);
  return acc;
}

// C/D fragment element (r) -> (row M, col N) within the 16x16 tile
static __device__ __forceinline__ void frag_rc(int r, int& mr, int& nc) {
  const int lane = threadIdx.x & 31;
  nc = lane & 15;
  mr = ((lane & 16) ? 8 : 0) + r;
}

// ---------------------------------------------------------------------------
__global__ void skvmn_f32_to_f16(const float* __restrict__ src,
                                 _Float16* __restrict__ dst, int n) {
  int i = blockIdx.x * blockDim.x + threadIdx.x;
  if (i < n) dst[i] = (_Float16)src[i];
}

__global__ void skvmn_init_mem(const float* __restrict__ Mv0,
                               float* __restrict__ memv) {
  int i = blockIdx.x * blockDim.x + threadIdx.x;
  if (i < B_TOT * M_MEM * D_DIM) memv[i] = Mv0[i % (M_MEM * D_DIM)];
}

// ---------------------------------------------------------------------------
__launch_bounds__(NTH, 1)
__global__ void skvmn_main(
    const int* __restrict__ qseq, const int* __restrict__ cseq,
    const float* __restrict__ emb_c, const float* __restrict__ emb_i,
    const float* __restrict__ bfv, const float* __restrict__ bav,
    const float* __restrict__ bev, const float* __restrict__ baddv,
    const float* __restrict__ bihv, const float* __restrict__ bhhv,
    const float* __restrict__ hx0, const float* __restrict__ cx0,
    const float* __restrict__ Wp, const float* __restrict__ bp,
    float* __restrict__ memv,
    _Float16* __restrict__ ftg, unsigned* __restrict__ sigg,
    _Float16* __restrict__ Hg,
    const _Float16* __restrict__ Mk16, const _Float16* __restrict__ Wf16,
    const _Float16* __restrict__ Wa16, const _Float16* __restrict__ We16,
    const _Float16* __restrict__ Wadd16, const _Float16* __restrict__ Wih16,
    const _Float16* __restrict__ Whh16,
    float* __restrict__ out) {
  // phase 3: gates in cols 0..511; phase 1: erase/add in cols 128..383, cw in 448..511
  __shared__ float    f32buf[BT][512];     // 32 KB
  __shared__ _Float16 aA[BT][256];         // 8 KB  concat A operand [read|k],[f|y]
  __shared__ _Float16 xb[BT][128];         // 4 KB  y_t / we / x_t staging
  __shared__ _Float16 hx16[BT][128];       // 4 KB  LSTM hidden (f16 A operand)
  __shared__ short    prevb[BT][S_LEN];    // 6.4 KB
  __shared__ float    lred[BT];

  const int tid   = threadIdx.x;
  const int wave  = tid >> 5;
  const int b0    = blockIdx.x * BT;
  const int brow  = tid >> 4;              // 0..15 (one batch row per 16 threads)
  const int dbase = (tid & 15) * 8;        // 0..120

  // ============================ Phase 1: DKVMN ============================
  for (int t = 0; t < S_LEN; ++t) {
    // gather k_t -> aA[:,128..255] (f16), y_t -> xb (f16)
    {
      const int q = qseq[(b0 + brow) * S_LEN + t];
      const int c = cseq[(b0 + brow) * S_LEN + t];
      const float* kp = emb_c + (size_t)q * D_DIM + dbase;
      const float* yp = emb_i + ((size_t)(q + NQ_C * c)) * D_DIM + dbase;
#pragma unroll
      for (int i = 0; i < 8; ++i) {
        aA[brow][128 + dbase + i] = (_Float16)kp[i];
        xb[brow][dbase + i]       = (_Float16)yp[i];
      }
    }
    __syncthreads();

    // cw logits: k_t[16,128] @ Mk^T[128,64]  (waves 0..3, one N-tile each)
    v8f accw = {0.f, 0.f, 0.f, 0.f, 0.f, 0.f, 0.f, 0.f};
    if (wave < 4)
      accw = gemm_tile(&aA[0][128], 256, Mk16 + wave * 16 * D_DIM, D_DIM, D_DIM);
    __syncthreads();
    if (wave < 4) {
#pragma unroll
      for (int r = 0; r < 8; ++r) {
        int mr, nc; frag_rc(r, mr, nc);
        f32buf[mr][448 + wave * 16 + nc] = accw[r];
      }
    }
    __syncthreads();

    // row softmax over M=64
    if (tid < BT) {
      float mx = -1e30f;
      for (int m = 0; m < M_MEM; ++m) mx = fmaxf(mx, f32buf[tid][448 + m]);
      float s = 0.f;
      for (int m = 0; m < M_MEM; ++m) {
        float e = __expf(f32buf[tid][448 + m] - mx);
        f32buf[tid][448 + m] = e; s += e;
      }
      const float inv = 1.f / s;
      for (int m = 0; m < M_MEM; ++m) f32buf[tid][448 + m] *= inv;
    }
    __syncthreads();

    // read = cw @ mem  -> f16 into aA[:,0..127]
    {
      float r[8] = {0, 0, 0, 0, 0, 0, 0, 0};
      const float* mp = memv + ((size_t)(b0 + brow) * M_MEM) * D_DIM + dbase;
      for (int m = 0; m < M_MEM; ++m) {
        const float c = f32buf[brow][448 + m];
        const float4 v0 = *(const float4*)(mp + (size_t)m * D_DIM);
        const float4 v1 = *(const float4*)(mp + (size_t)m * D_DIM + 4);
        r[0] += c * v0.x; r[1] += c * v0.y; r[2] += c * v0.z; r[3] += c * v0.w;
        r[4] += c * v1.x; r[5] += c * v1.y; r[6] += c * v1.z; r[7] += c * v1.w;
      }
#pragma unroll
      for (int i = 0; i < 8; ++i) aA[brow][dbase + i] = (_Float16)r[i];
    }
    __syncthreads();

    // f = tanh([read,k] @ Wf^T + bf); store f16 -> aA[:,0..127] and ft (global)
    {
      v8f acc = gemm_tile(&aA[0][0], 256, Wf16 + wave * 16 * 256, 256, 256);
      __syncthreads();                       // all reads of aA done before rewrite
#pragma unroll
      for (int r = 0; r < 8; ++r) {
        int mr, nc; frag_rc(r, mr, nc);
        const int n = wave * 16 + nc;
        const float v = tanhf(acc[r] + bfv[n]);
        aA[mr][n] = (_Float16)v;
        ftg[((size_t)t * B_TOT + b0 + mr) * D_DIM + n] = (_Float16)v;
      }
#pragma unroll
      for (int i = 0; i < 8; ++i)            // y_t -> aA[:,128..255]
        aA[brow][128 + dbase + i] = xb[brow][dbase + i];
    }
    __syncthreads();

    // we = [f,y] @ Wa^T + ba -> f16 into xb
    {
      v8f acc = gemm_tile(&aA[0][0], 256, Wa16 + wave * 16 * 256, 256, 256);
      __syncthreads();
#pragma unroll
      for (int r = 0; r < 8; ++r) {
        int mr, nc; frag_rc(r, mr, nc);
        const int n = wave * 16 + nc;
        xb[mr][n] = (_Float16)(acc[r] + bav[n]);
      }
    }
    __syncthreads();

    // erase = sigmoid(we@We^T+be) -> cols 128..255 ; add = tanh(we@Wadd^T+badd) -> cols 256..383
    {
      v8f ae = gemm_tile(&xb[0][0], 128, We16 + wave * 16 * 128, 128, 128);
      v8f aa = gemm_tile(&xb[0][0], 128, Wadd16 + wave * 16 * 128, 128, 128);
#pragma unroll
      for (int r = 0; r < 8; ++r) {
        int mr, nc; frag_rc(r, mr, nc);
        const int n = wave * 16 + nc;
        f32buf[mr][128 + n] = sigm(ae[r] + bev[n]);
        f32buf[mr][256 + n] = tanhf(aa[r] + baddv[n]);
      }
    }
    __syncthreads();

    // mem = mem*(1 - cw (x) erase) + cw (x) add
    {
      float* mp = memv + ((size_t)(b0 + brow) * M_MEM) * D_DIM + dbase;
      float er[8], ad[8];
#pragma unroll
      for (int i = 0; i < 8; ++i) {
        er[i] = f32buf[brow][128 + dbase + i];
        ad[i] = f32buf[brow][256 + dbase + i];
      }
      for (int m = 0; m < M_MEM; ++m) {
        const float c = f32buf[brow][448 + m];
        float4 v0 = *(const float4*)(mp + (size_t)m * D_DIM);
        float4 v1 = *(const float4*)(mp + (size_t)m * D_DIM + 4);
        v0.x = v0.x * (1.f - c * er[0]) + c * ad[0];
        v0.y = v0.y * (1.f - c * er[1]) + c * ad[1];
        v0.z = v0.z * (1.f - c * er[2]) + c * ad[2];
        v0.w = v0.w * (1.f - c * er[3]) + c * ad[3];
        v1.x = v1.x * (1.f - c * er[4]) + c * ad[4];
        v1.y = v1.y * (1.f - c * er[5]) + c * ad[5];
        v1.z = v1.z * (1.f - c * er[6]) + c * ad[6];
        v1.w = v1.w * (1.f - c * er[7]) + c * ad[7];
        *(float4*)(mp + (size_t)m * D_DIM)     = v0;
        *(float4*)(mp + (size_t)m * D_DIM + 4) = v1;
      }
    }
    // triangular membership -> 2-bit identity codes packed to 128-bit signature
    if (tid < BT) {
      unsigned sg[4] = {0, 0, 0, 0};
      for (int m = 0; m < M_MEM; ++m) {
        const float w = f32buf[tid][448 + m];
        float tri = fminf((w - 0.01f) * 25.f, (0.1f - w) * 20.f);
        tri = fmaxf(tri, 0.f);
        const unsigned iv = (tri >= 0.6f) ? 2u : ((tri >= 0.1f) ? 1u : 0u);
        sg[m >> 4] |= iv << ((m & 15) * 2);
      }
      unsigned* d = sigg + ((size_t)(b0 + tid) * S_LEN + t) * 4;
      d[0] = sg[0]; d[1] = sg[1]; d[2] = sg[2]; d[3] = sg[3];
    }
    __syncthreads();
  }

  // ====== Phase 2: prev_t = max{ j<t : sig[j]==sig[t] }  (else -1) ======
  for (int task = tid; task < BT * S_LEN; task += NTH) {
    const int b = task / S_LEN, t = task % S_LEN;
    const unsigned* sb = sigg + (size_t)(b0 + b) * S_LEN * 4;
    const unsigned s0 = sb[t * 4], s1 = sb[t * 4 + 1];
    const unsigned s2 = sb[t * 4 + 2], s3 = sb[t * 4 + 3];
    short pv = -1;
    for (int j = t - 1; j >= 0; --j) {
      if (sb[j * 4] == s0 && sb[j * 4 + 1] == s1 &&
          sb[j * 4 + 2] == s2 && sb[j * 4 + 3] == s3) { pv = (short)j; break; }
    }
    prevb[b][t] = pv;
  }

  // init hx (shared f16) and cx (registers, 8 per thread)
  float cxr[8];
#pragma unroll
  for (int i = 0; i < 8; ++i) {
    hx16[brow][dbase + i] = (_Float16)hx0[dbase + i];
    cxr[i] = cx0[dbase + i];
  }
  // warm L2 with LSTM weights (global_prefetch_b8)
  for (size_t o = (size_t)tid * 256; o < (size_t)N_WIH16 * 2; o += (size_t)NTH * 256) {
    __builtin_prefetch((const char*)Wih16 + o, 0, 1);
    __builtin_prefetch((const char*)Whh16 + o, 0, 1);
  }
  __syncthreads();

  // ============================ Phase 3: hop-LSTM =========================
  for (int t = 0; t < S_LEN; ++t) {
    // stage x_t (f16) into xb; substitute hx rows from H[prev]
    {
      const uint4 v = *(const uint4*)((const char*)ftg +
                        ((size_t)t * B_TOT + b0) * D_DIM * 2 + (size_t)tid * 16);
      *(uint4*)((char*)&xb[0][0] + (size_t)tid * 16) = v;
      const short pv = prevb[brow][t];
      if (pv >= 0) {
        const _Float16* hs = Hg + ((size_t)pv * B_TOT + b0 + brow) * D_DIM + dbase;
#pragma unroll
        for (int i = 0; i < 8; ++i) hx16[brow][dbase + i] = hs[i];
      }
    }
    if (tid < BT) lred[tid] = bp[0];
    __syncthreads();

    // gates[16,512] = x @ W_ih^T + hx @ W_hh^T + b_ih + b_hh
#pragma unroll
    for (int nt = 0; nt < 4; ++nt) {
      const int n0 = wave * 4 + nt;                      // 32 N-tiles over 8 waves
      v8f acc = gemm_tile(&xb[0][0], 128, Wih16 + (size_t)n0 * 16 * 128, 128, 128);
      for (int k = 0; k < 128; k += 32)
        acc = wmma32(frag_ld(&hx16[0][0] + k, 128),
                     frag_ld(Whh16 + (size_t)n0 * 16 * 128 + k, 128), acc);
#pragma unroll
      for (int r = 0; r < 8; ++r) {
        int mr, nc; frag_rc(r, mr, nc);
        const int n = n0 * 16 + nc;
        f32buf[mr][n] = acc[r] + bihv[n] + bhhv[n];
      }
    }
    __syncthreads();

    // elementwise LSTM + logit partial reduction
    {
      float part = 0.f;
#pragma unroll
      for (int i = 0; i < 8; ++i) {
        const int d = dbase + i;
        const float gi = f32buf[brow][d];
        const float gf = f32buf[brow][128 + d];
        const float gg = f32buf[brow][256 + d];
        const float go = f32buf[brow][384 + d];
        cxr[i] = sigm(gf) * cxr[i] + sigm(gi) * tanhf(gg);
        const float h = sigm(go) * tanhf(cxr[i]);
        hx16[brow][d] = (_Float16)h;
        Hg[((size_t)t * B_TOT + b0 + brow) * D_DIM + d] = (_Float16)h;
        part += h * Wp[d];
      }
      atomicAdd(&lred[brow], part);
    }
    __syncthreads();
    if (tid < BT)
      out[(size_t)(b0 + tid) * S_LEN + t] = sigm(lred[tid]);
    __syncthreads();
  }
}

// ---------------------------------------------------------------------------
extern "C" void kernel_launch(void* const* d_in, const int* in_sizes, int n_in,
                              void* d_out, int out_size, void* d_ws, size_t ws_size,
                              hipStream_t stream) {
  (void)in_sizes; (void)n_in; (void)out_size; (void)ws_size;
  const int*   qseq = (const int*)d_in[0];
  const int*   cseq = (const int*)d_in[1];
  const float* Mk   = (const float*)d_in[2];
  const float* Mv0  = (const float*)d_in[3];
  const float* embc = (const float*)d_in[4];
  const float* embi = (const float*)d_in[5];
  const float* Wf   = (const float*)d_in[6];
  const float* bf   = (const float*)d_in[7];
  const float* Wa   = (const float*)d_in[8];
  const float* ba   = (const float*)d_in[9];
  const float* We   = (const float*)d_in[10];
  const float* be   = (const float*)d_in[11];
  const float* Wadd = (const float*)d_in[12];
  const float* badd = (const float*)d_in[13];
  const float* Wih  = (const float*)d_in[14];
  const float* Whh  = (const float*)d_in[15];
  const float* bih  = (const float*)d_in[16];
  const float* bhh  = (const float*)d_in[17];
  const float* hx0  = (const float*)d_in[18];
  const float* cx0  = (const float*)d_in[19];
  const float* Wp   = (const float*)d_in[20];
  const float* bp   = (const float*)d_in[21];
  float* out = (float*)d_out;

  char* ws = (char*)d_ws;
  size_t off = 0;
  auto take = [&](size_t bytes) {
    char* r = ws + off;
    off = (off + bytes + 255) & ~(size_t)255;
    return r;
  };
  float*     memv   = (float*)take(WS_MEM_B);
  _Float16*  ftg    = (_Float16*)take(WS_FT_B);
  unsigned*  sigg   = (unsigned*)take(WS_SIG_B);
  _Float16*  Hg     = (_Float16*)take(WS_H_B);
  _Float16*  Mk16   = (_Float16*)take((size_t)N_MK16 * 2);
  _Float16*  Wf16   = (_Float16*)take((size_t)N_WF16 * 2);
  _Float16*  Wa16   = (_Float16*)take((size_t)N_WA16 * 2);
  _Float16*  We16   = (_Float16*)take((size_t)N_WE16 * 2);
  _Float16*  Wadd16 = (_Float16*)take((size_t)N_WADD16 * 2);
  _Float16*  Wih16  = (_Float16*)take((size_t)N_WIH16 * 2);
  _Float16*  Whh16  = (_Float16*)take((size_t)N_WHH16 * 2);

  auto cvt = [&](const float* s, _Float16* d, int n) {
    skvmn_f32_to_f16<<<(n + 255) / 256, 256, 0, stream>>>(s, d, n);
  };
  cvt(Mk, Mk16, N_MK16);
  cvt(Wf, Wf16, N_WF16);
  cvt(Wa, Wa16, N_WA16);
  cvt(We, We16, N_WE16);
  cvt(Wadd, Wadd16, N_WADD16);
  cvt(Wih, Wih16, N_WIH16);
  cvt(Whh, Whh16, N_WHH16);
  skvmn_init_mem<<<(B_TOT * M_MEM * D_DIM + 255) / 256, 256, 0, stream>>>(Mv0, memv);

  skvmn_main<<<B_TOT / BT, NTH, 0, stream>>>(
      qseq, cseq, embc, embi, bf, ba, be, badd, bih, bhh, hx0, cx0, Wp, bp,
      memv, ftg, sigg, Hg, Mk16, Wf16, Wa16, We16, Wadd16, Wih16, Whh16, out);
}
